// TransformerGDMOE_25142738550854
// MI455X (gfx1250) — matrix operand order repaired and checked by reading, not compiled, not run on hardware
//
#include <hip/hip_runtime.h>
#include <math.h>

// ---------------- problem constants ----------------
constexpr int kIn   = 128;            // INPUT_SIZE
constexpr int kEmb  = 512;            // EMB
constexpr int kD    = kIn + 3*kEmb;   // 1664
constexpr int kN    = 1024;           // N
constexpr int kB    = 4;              // B
constexpr int kCtx  = kN - 1;         // 1023 (NQ=1)
constexpr int kH    = 4;              // heads for alpha
constexpr int kM    = 8;              // experts
constexpr int kHid  = 1024;           // HID
constexpr int kL    = 2;              // layers
constexpr int kTok  = kB * kN;        // 4096 tokens
constexpr int kMaxPairs = kTok;       // worst case: every token picks this expert
#define SOFT_SCALE (1.0e10f / 40.79215674f)   // LAM^2 / sqrt(D)

#if defined(__has_builtin)
#if __has_builtin(__builtin_amdgcn_tensor_load_to_lds)
#define HAVE_TDM 1
#endif
#endif
#ifndef HAVE_TDM
#define HAVE_TDM 0
#endif

// ---------------- bf16 helpers + WMMA ----------------
typedef __attribute__((ext_vector_type(16))) __bf16 v16bf;
typedef __attribute__((ext_vector_type(8)))  float  v8f;
typedef __attribute__((ext_vector_type(4)))  unsigned int v4u;
typedef __attribute__((ext_vector_type(8)))  int v8i;
typedef __attribute__((ext_vector_type(4)))  int v4i;

union BF16Frag { v16bf v; __bf16 e[16]; unsigned short s[16]; };

__device__ __forceinline__ unsigned short f2bf(float f) {
  union { __bf16 b; unsigned short s; } u; u.b = (__bf16)f; return u.s;
}
__device__ __forceinline__ float bf2f(unsigned short s) {
  union { unsigned u; float f; } x; x.u = ((unsigned)s) << 16;
  return x.f;
}
// ISA 7.12.2: 16-bit A (16x32): lanes 0-15 -> M, VGPR0-3 K={0..7|8..15}, VGPR4-7 K={16..23|24..31}
__device__ __forceinline__ int a_k(int lane, int h) {
  int kb = (lane >> 4) * 8;
  return (h < 8) ? (kb + h) : (16 + kb + (h - 8));
}
// B (32x16): lanes 0-15 -> N, same K order as A: K={0..7|8..15} then K={16..23|24..31}
__device__ __forceinline__ int b_k(int lane, int h) {
  int kb = (lane >> 4) * 8;
  return (h < 8) ? (kb + h) : (16 + kb + (h - 8));
}
__device__ __forceinline__ v8f wmma_bf16(v16bf a, v16bf b, v8f c) {
  return __builtin_amdgcn_wmma_f32_16x16x32_bf16(false, a, false, b, (short)0, c, false, false);
}
__device__ __forceinline__ float gelu_exact(float x) {
  return 0.5f * x * (1.0f + erff(x * 0.70710678118f));
}

#if HAVE_TDM
// TDM gather: load `nvalid` rows (row length `row_elems` bf16) of a 2D tensor
// (`tensor_rows` x `row_elems`, contiguous) into LDS at `lds_off`, rows packed
// consecutively.  D# per CDNA5 ISA ch.8: gather_mode=1, 16-bit indices,
// data_size=1 (2 bytes), type=2.  Issue from ONE wave; TENSORcnt waited here.
// amdgpu-toolchain (clang-23) builtin arity is 6:
//   (v4u g0, v8i g1, v4i g2, v4i g3, v8i extra, i32 cpol)
__device__ __forceinline__ void tdm_gather_rows(unsigned lds_off, const void* base,
                                                unsigned tensor_rows, unsigned row_elems,
                                                const int* rowIdx, int nvalid) {
  unsigned long long ga = (unsigned long long)base;
  v4u g0;
  g0.x = 0x80000001u;                                  // count=1 | gather_mode | 16b idx
  g0.y = lds_off;                                      // lds_addr (bytes)
  g0.z = (unsigned)(ga & 0xFFFFFFFFu);                 // global_addr[31:0]
  g0.w = (unsigned)((ga >> 32) & 0x01FFFFFFu) | (2u << 30);   // global_addr[56:32] | type=2
  v8i g1;
  g1[0] = 0x00010000;                                  // data_size=1 (2 bytes)
  g1[1] = (int)((row_elems & 0xFFFFu) << 16);          // tensor_dim0[15:0]
  g1[2] = (int)((row_elems >> 16) | ((tensor_rows & 0xFFFFu) << 16)); // dim0 hi | dim1 lo
  g1[3] = (int)((tensor_rows >> 16) | ((row_elems & 0xFFFFu) << 16)); // dim1 hi | tile_dim0
  g1[4] = nvalid & 0xFFFF;                             // tile_dim1 = #valid indices
  g1[5] = (int)row_elems;                              // tensor_dim0_stride lo
  g1[6] = 0;
  g1[7] = 0;
  v4i g2, g3;
#pragma unroll
  for (int j = 0; j < 4; ++j) {
    g2[j] = __builtin_amdgcn_readfirstlane((rowIdx[2*j]     & 0xFFFF) | (rowIdx[2*j + 1]  << 16));
    g3[j] = __builtin_amdgcn_readfirstlane((rowIdx[8 + 2*j] & 0xFFFF) | (rowIdx[9 + 2*j] << 16));
  }
  v8i gx = {0, 0, 0, 0, 0, 0, 0, 0};                   // extra group (unused, zero-fill)
  __builtin_amdgcn_tensor_load_to_lds(g0, g1, g2, g3, gx, 0);
  __builtin_amdgcn_s_wait_tensorcnt((short)0);
}
#endif

// ---------------- K0: weights -> bf16 (once, both layers) ----------------
__global__ void cvt_weights_kernel(const float* __restrict__ W1, const float* __restrict__ W2,
                                   unsigned short* __restrict__ W1b, unsigned short* __restrict__ W2b,
                                   size_t n) {
  size_t i = (size_t)blockIdx.x * blockDim.x + threadIdx.x;
  if (i < n) { W1b[i] = f2bf(W1[i]); W2b[i] = f2bf(W2[i]); }
}

// ---------------- K1: alpha_eff[e] = sum_h kp[h]*alpha[layer,h,e] ----------------
__global__ void alpha_eff_kernel(const float* __restrict__ alpha, const float* __restrict__ kp,
                                 float* __restrict__ ae, int layer) {
  int e = threadIdx.x;
  float s = 0.f;
  for (int h = 0; h < kH; ++h) s += kp[h] * alpha[((size_t)layer * kH + h) * kEmb + e];
  ae[e] = s;
}

// ------- K2: S[b,m,n] = sum_i h[b,i,m]*h[b,i,n]; f32 for softmax + bf16 copy for attn
__global__ void s_gemm_kernel(const float* __restrict__ hh, float* __restrict__ S,
                              unsigned short* __restrict__ Sb16) {
  const int lane = threadIdx.x & 31, wave = threadIdx.x >> 5;
  const int m0 = blockIdx.x * 16;
  const int n0 = (blockIdx.y * 4 + wave) * 16;
  const int b  = blockIdx.z;
  const float* hb = hh + (size_t)b * kD * kN;
  const int mL = lane & 15;
  v8f acc = {};
  for (int i0 = 0; i0 < kIn; i0 += 32) {
    BF16Frag a, bb;
#pragma unroll
    for (int t = 0; t < 16; ++t) {
      a.e[t]  = (__bf16)hb[(size_t)(i0 + a_k(lane, t)) * kN + m0 + mL];   // A[m][i]=h[i][m]
      bb.e[t] = (__bf16)hb[(size_t)(i0 + b_k(lane, t)) * kN + n0 + mL];   // B[i][n]=h[i][n]
    }
    acc = wmma_bf16(a.v, bb.v, acc);
  }
  float*          Sb = S    + (size_t)b * kN * kN;
  unsigned short* Sh = Sb16 + (size_t)b * kN * kN;
#pragma unroll
  for (int r = 0; r < 8; ++r) {
    int row = m0 + r + ((lane >> 4) << 3);
    int col = n0 + mL;
    float v = (col == kCtx) ? 0.0f : acc[r];
    Sb[(size_t)row * kN + col] = v;
    Sh[(size_t)row * kN + col] = f2bf(v);
  }
}

// ---------------- K3: S2 = softmax(SOFT_SCALE*S) over n<1023, bf16 out; S2[1023]=0 --
__global__ void softmax_kernel(const float* __restrict__ S, unsigned short* __restrict__ S2) {
  __shared__ float red[256];
  const float*    src = S  + (size_t)blockIdx.x * kN;
  unsigned short* dst = S2 + (size_t)blockIdx.x * kN;
  float mx = -INFINITY;
  for (int n = threadIdx.x; n < kCtx; n += 256) mx = fmaxf(mx, src[n]);
  red[threadIdx.x] = mx; __syncthreads();
  for (int s = 128; s > 0; s >>= 1) { if (threadIdx.x < s) red[threadIdx.x] = fmaxf(red[threadIdx.x], red[threadIdx.x + s]); __syncthreads(); }
  mx = red[0]; __syncthreads();
  float sum = 0.f;
  for (int n = threadIdx.x; n < kCtx; n += 256) sum += __expf(SOFT_SCALE * (src[n] - mx));
  red[threadIdx.x] = sum; __syncthreads();
  for (int s = 128; s > 0; s >>= 1) { if (threadIdx.x < s) red[threadIdx.x] += red[threadIdx.x + s]; __syncthreads(); }
  float inv = 1.0f / red[0];
  for (int n = threadIdx.x; n < kCtx; n += 256) dst[n] = f2bf(__expf(SOFT_SCALE * (src[n] - mx)) * inv);
  if (threadIdx.x == 0) dst[kCtx] = 0;
}

// ------- K4: accE = delta@S^T, accM = v2@S2^T ; h[1152+e,m] += accE/1023 ; AM=accM/1023
__global__ void attn_kernel(float* __restrict__ hh, const unsigned short* __restrict__ Sb16,
                            const unsigned short* __restrict__ S2, const float* __restrict__ ae,
                            float* __restrict__ AM) {
  const int lane = threadIdx.x & 31, wave = threadIdx.x >> 5;
  const int e0 = blockIdx.x * 16;
  const int m0 = (blockIdx.y * 4 + wave) * 16;
  const int b  = blockIdx.z;
  const float* hb = hh + (size_t)b * kD * kN;
  const unsigned short* Sb  = Sb16 + (size_t)b * kN * kN;
  const unsigned short* S2b = S2   + (size_t)b * kN * kN;
  const int eL = lane & 15;
  const float aev = ae[e0 + eL];
  const float* v1row = hb + (size_t)(kIn + e0 + eL) * kN;
  const float* v2row = hb + (size_t)(kIn + kEmb + e0 + eL) * kN;
  v8f accE = {}, accM = {};
  for (int n0 = 0; n0 < kN; n0 += 32) {
    BF16Frag aD, aV, bS, bS2;
#pragma unroll
    for (int t = 0; t < 16; ++t) {
      int ka = a_k(lane, t);
      float v1 = v1row[n0 + ka], v2 = v2row[n0 + ka];
      aD.e[t] = (__bf16)(aev * (v1 - v2));
      aV.e[t] = (__bf16)v2;
      int kb = b_k(lane, t);                       // B[n][m] = S[m][n]
      bS.s[t]  = Sb [(size_t)(m0 + eL) * kN + n0 + kb];
      bS2.s[t] = S2b[(size_t)(m0 + eL) * kN + n0 + kb];
    }
    accE = wmma_bf16(aD.v, bS.v,  accE);
    accM = wmma_bf16(aV.v, bS2.v, accM);
  }
  const float inv = 1.0f / (float)kCtx;
#pragma unroll
  for (int r = 0; r < 8; ++r) {
    int er  = e0 + r + ((lane >> 4) << 3);
    int col = m0 + eL;
    hh[(size_t)b * kD * kN + (size_t)(kD - kEmb + er) * kN + col] += accE[r] * inv;
    AM[((size_t)b * kEmb + er) * kN + col] = accM[r] * inv;
  }
}

// ---------------- K5: h[b,640+e,n] -= AM[b,e,n] ----------------
__global__ void apply_mid_kernel(float* __restrict__ hh, const float* __restrict__ AM) {
  size_t i = (size_t)blockIdx.x * 256 + threadIdx.x;        // < B*EMB*N
  int b = (int)(i / ((size_t)kEmb * kN));
  int rem = (int)(i % ((size_t)kEmb * kN));
  int e = rem / kN, n = rem % kN;
  hh[(size_t)b * kD * kN + (size_t)(kIn + kEmb + e) * kN + n] -= AM[i];
}

// ---------------- K6: Ht[t,k] = bf16(h[b,1152+k,n]) token-major ----------------
__global__ void hs_bf16_kernel(const float* __restrict__ hh, unsigned short* __restrict__ Ht) {
  int t = blockIdx.x * 256 + threadIdx.x;
  int b = t >> 10, n = t & (kN - 1);
  const float* base = hh + (size_t)b * kD * kN + (size_t)(kD - kEmb) * kN + n;
  unsigned short* dst = Ht + (size_t)t * kEmb;
  for (int k = 0; k < kEmb; ++k) dst[k] = f2bf(base[(size_t)k * kN]);
}

__global__ void zero_counts_kernel(int* counts) { if (threadIdx.x < kM) counts[threadIdx.x] = 0; }

// ---------------- K7: gating (wave per token), top-2 of 8, build expert lists ------
__global__ void gate_kernel(const unsigned short* __restrict__ Ht, const float* __restrict__ xi,
                            float* __restrict__ gateOf, int* __restrict__ counts,
                            int* __restrict__ lists) {
  const int lane = threadIdx.x & 31, wave = threadIdx.x >> 5;
  const int t = blockIdx.x * 4 + wave;
  const unsigned short* hrow = Ht + (size_t)t * kEmb;
  float a[kM];
#pragma unroll
  for (int m = 0; m < kM; ++m) {
    float p = 0.f;
    for (int k = lane; k < kEmb; k += 32) p += bf2f(hrow[k]) * xi[m * kEmb + k];
    for (int off = 16; off > 0; off >>= 1) p += __shfl_xor(p, off, 32);
    a[m] = p;
  }
  if (lane == 0) {
    float mx = a[0];
    for (int m = 1; m < kM; ++m) mx = fmaxf(mx, a[m]);
    float ex[kM], s = 0.f;
    for (int m = 0; m < kM; ++m) { ex[m] = __expf(a[m] - mx); s += ex[m]; }
    int i0 = 0;
    for (int m = 1; m < kM; ++m) if (a[m] > a[i0]) i0 = m;     // first max (tie = first)
    int i1 = -1;
    for (int m = 0; m < kM; ++m) if (m != i0 && (i1 < 0 || a[m] > a[i1])) i1 = m;
    float inv = 1.0f / s;
    gateOf[2 * t]     = ex[i0] * inv;
    gateOf[2 * t + 1] = ex[i1] * inv;
    int p0 = atomicAdd(&counts[i0], 1); lists[i0 * kMaxPairs + p0] = 2 * t;
    int p1 = atomicAdd(&counts[i1], 1); lists[i1 * kMaxPairs + p1] = 2 * t + 1;
  }
}

// ---------------- K8: grouped GEMM stage 1: Z = gelu(Hs@W1 + b1), bf16 out --------
__global__ void ffn1_kernel(const unsigned short* __restrict__ Ht,
                            const unsigned short* __restrict__ W1b, const float* __restrict__ b1,
                            const int* __restrict__ counts, const int* __restrict__ lists,
                            unsigned short* __restrict__ Zb, int layer) {
  __shared__ unsigned short As[16 * kEmb];   // 16 KB
  __shared__ int pairIdx[16], rowValid[16], rowIdx[16];
  const int e = blockIdx.z, rt = blockIdx.x;
  const int cnt = counts[e];
  if (rt * 16 >= cnt) return;
  if (threadIdx.x < 16) {
    int i = rt * 16 + threadIdx.x;
    int valid = i < cnt;
    int p = valid ? lists[e * kMaxPairs + i] : lists[e * kMaxPairs];
    pairIdx[threadIdx.x] = p;
    rowValid[threadIdx.x] = valid;
    rowIdx[threadIdx.x] = p >> 1;              // token row in Ht
  }
  __syncthreads();
#if HAVE_TDM
  if (threadIdx.x < 32) {                      // wave 0 drives the Tensor Data Mover
    int nvalid = min(16, cnt - rt * 16);
    tdm_gather_rows((unsigned)(size_t)(void*)As, Ht, kTok, kEmb, rowIdx, nvalid);
  }
  __syncthreads();
#else
  for (int idx = threadIdx.x; idx < 16 * (kEmb / 2); idx += blockDim.x) {
    int row = idx / (kEmb / 2), c2 = idx % (kEmb / 2);
    ((unsigned*)As)[row * (kEmb / 2) + c2] =
        ((const unsigned*)(Ht + (size_t)rowIdx[row] * kEmb))[c2];
  }
  __syncthreads();
#endif
  const int lane = threadIdx.x & 31, wave = threadIdx.x >> 5;
  const int hbase = blockIdx.y * 128 + wave * 32;
  const unsigned short* Wb = W1b + ((size_t)layer * kM + e) * kEmb * kHid;
  const int mL = lane & 15;
  v8f acc0 = {}, acc1 = {};
  for (int k0 = 0; k0 < kEmb; k0 += 32) {
    BF16Frag a, b0, b1f;
#pragma unroll
    for (int t = 0; t < 16; ++t) {
      a.s[t] = As[mL * kEmb + k0 + a_k(lane, t)];
      int kb = b_k(lane, t);
      b0.s[t]  = Wb[(size_t)(k0 + kb) * kHid + hbase + mL];
      b1f.s[t] = Wb[(size_t)(k0 + kb) * kHid + hbase + 16 + mL];
    }
    acc0 = wmma_bf16(a.v, b0.v,  acc0);
    acc1 = wmma_bf16(a.v, b1f.v, acc1);
  }
  const float* bias = b1 + ((size_t)layer * kM + e) * kHid;
#pragma unroll
  for (int r = 0; r < 8; ++r) {
    int row = r + ((lane >> 4) << 3);
    if (!rowValid[row]) continue;
    size_t zrow = (size_t)pairIdx[row] * kHid;
    int c0 = hbase + mL, c1 = hbase + 16 + mL;
    Zb[zrow + c0] = f2bf(gelu_exact(acc0[r] + bias[c0]));
    Zb[zrow + c1] = f2bf(gelu_exact(acc1[r] + bias[c1]));
  }
}

// ---------------- K9: stage 2: O = gate * (Z@W2 + b2), fp32 out -------------------
__global__ void ffn2_kernel(const unsigned short* __restrict__ Zb,
                            const unsigned short* __restrict__ W2b, const float* __restrict__ b2,
                            const float* __restrict__ gateOf,
                            const int* __restrict__ counts, const int* __restrict__ lists,
                            float* __restrict__ O, int layer) {
  __shared__ unsigned short As[16 * kHid];   // 32 KB
  __shared__ int pairIdx[16], rowValid[16];
  const int e = blockIdx.z, rt = blockIdx.x;
  const int cnt = counts[e];
  if (rt * 16 >= cnt) return;
  if (threadIdx.x < 16) {
    int i = rt * 16 + threadIdx.x;
    int valid = i < cnt;
    pairIdx[threadIdx.x] = valid ? lists[e * kMaxPairs + i] : lists[e * kMaxPairs];
    rowValid[threadIdx.x] = valid;
  }
  __syncthreads();
#if HAVE_TDM
  if (threadIdx.x < 32) {                      // wave 0 drives the Tensor Data Mover
    int nvalid = min(16, cnt - rt * 16);
    tdm_gather_rows((unsigned)(size_t)(void*)As, Zb, 2 * kTok, kHid, pairIdx, nvalid);
  }
  __syncthreads();
#else
  for (int idx = threadIdx.x; idx < 16 * (kHid / 2); idx += blockDim.x) {
    int row = idx / (kHid / 2), c2 = idx % (kHid / 2);
    ((unsigned*)As)[row * (kHid / 2) + c2] =
        ((const unsigned*)(Zb + (size_t)pairIdx[row] * kHid))[c2];
  }
  __syncthreads();
#endif
  const int lane = threadIdx.x & 31, wave = threadIdx.x >> 5;
  const int cbase = blockIdx.y * 128 + wave * 32;
  const unsigned short* Wb = W2b + ((size_t)layer * kM + e) * kHid * kEmb;
  const int mL = lane & 15;
  v8f acc0 = {}, acc1 = {};
  for (int k0 = 0; k0 < kHid; k0 += 32) {
    BF16Frag a, b0, b1f;
#pragma unroll
    for (int t = 0; t < 16; ++t) {
      a.s[t] = As[mL * kHid + k0 + a_k(lane, t)];
      int kb = b_k(lane, t);
      b0.s[t]  = Wb[(size_t)(k0 + kb) * kEmb + cbase + mL];
      b1f.s[t] = Wb[(size_t)(k0 + kb) * kEmb + cbase + 16 + mL];
    }
    acc0 = wmma_bf16(a.v, b0.v,  acc0);
    acc1 = wmma_bf16(a.v, b1f.v, acc1);
  }
  const float* bias = b2 + ((size_t)layer * kM + e) * kEmb;
#pragma unroll
  for (int r = 0; r < 8; ++r) {
    int row = r + ((lane >> 4) << 3);
    if (!rowValid[row]) continue;
    float g = gateOf[pairIdx[row]];
    int c0 = cbase + mL, c1 = cbase + 16 + mL;
    O[(size_t)pairIdx[row] * kEmb + c0] = g * (acc0[r] + bias[c0]);
    O[(size_t)pairIdx[row] * kEmb + c1] = g * (acc1[r] + bias[c1]);
  }
}

// ---------------- K10: h[b,1152+k,n] += O[2t,k] + O[2t+1,k]  (deterministic) -------
__global__ void combine_kernel(float* __restrict__ hh, const float* __restrict__ O) {
  size_t i = (size_t)blockIdx.x * 256 + threadIdx.x;   // < kTok*kEmb
  int t = (int)(i / kEmb), k = (int)(i % kEmb);
  int b = t >> 10, n = t & (kN - 1);
  float add = O[(size_t)(2 * t) * kEmb + k] + O[(size_t)(2 * t + 1) * kEmb + k];
  hh[(size_t)b * kD * kN + (size_t)(kD - kEmb + k) * kN + n] += add;
}

// ------------------------------- launcher -----------------------------------------
extern "C" void kernel_launch(void* const* d_in, const int* in_sizes, int n_in,
                              void* d_out, int out_size, void* d_ws, size_t ws_size,
                              hipStream_t stream) {
  const float* h_in  = (const float*)d_in[0];
  const float* alpha = (const float*)d_in[1];
  const float* kp    = (const float*)d_in[2];
  const float* xi    = (const float*)d_in[3];
  const float* W1    = (const float*)d_in[4];
  const float* b1    = (const float*)d_in[5];
  const float* W2    = (const float*)d_in[6];
  const float* b2    = (const float*)d_in[7];
  float* h = (float*)d_out;

  // workspace carve-out (~110 MB)
  char* w = (char*)d_ws;
  auto carve = [&](size_t bytes) -> char* { char* p = w; w += (bytes + 255) & ~(size_t)255; return p; };
  const size_t nW = (size_t)kL * kM * kEmb * kHid;                 // 8,388,608
  unsigned short* W1b = (unsigned short*)carve(nW * 2);
  unsigned short* W2b = (unsigned short*)carve(nW * 2);
  float*          S    = (float*)carve((size_t)kB * kN * kN * 4);
  unsigned short* Sb16 = (unsigned short*)carve((size_t)kB * kN * kN * 2);
  unsigned short* S2   = (unsigned short*)carve((size_t)kB * kN * kN * 2);
  float* AM  = (float*)carve((size_t)kB * kEmb * kN * 4);
  float* ae  = (float*)carve(kEmb * 4);
  unsigned short* Ht = (unsigned short*)carve((size_t)kTok * kEmb * 2);
  int* counts = (int*)carve(kM * 4);
  int* lists  = (int*)carve((size_t)kM * kMaxPairs * 4);
  float* gateOf = (float*)carve((size_t)2 * kTok * 4);
  unsigned short* Zb = (unsigned short*)carve((size_t)2 * kTok * kHid * 2);
  float* O = (float*)carve((size_t)2 * kTok * kEmb * 4);
  (void)ws_size; (void)n_in; (void)in_sizes;

  // h (in-place working copy) = input h
  (void)hipMemcpyAsync(h, h_in, (size_t)out_size * sizeof(float), hipMemcpyDeviceToDevice, stream);

  // weights -> bf16 once
  cvt_weights_kernel<<<(unsigned)((nW + 255) / 256), 256, 0, stream>>>(W1, W2, W1b, W2b, nW);

  for (int layer = 0; layer < kL; ++layer) {
    alpha_eff_kernel<<<1, kEmb, 0, stream>>>(alpha, kp, ae, layer);
    s_gemm_kernel<<<dim3(kN / 16, 16, kB), 128, 0, stream>>>(h, S, Sb16);
    softmax_kernel<<<kB * kN, 256, 0, stream>>>(S, S2);
    attn_kernel<<<dim3(kEmb / 16, 16, kB), 128, 0, stream>>>(h, Sb16, S2, ae, AM);
    apply_mid_kernel<<<(kB * kEmb * kN) / 256, 256, 0, stream>>>(h, AM);
    hs_bf16_kernel<<<kTok / 256, 256, 0, stream>>>(h, Ht);
    zero_counts_kernel<<<1, 32, 0, stream>>>(counts);
    gate_kernel<<<kTok / 4, 128, 0, stream>>>(Ht, xi, gateOf, counts, lists);
    ffn1_kernel<<<dim3(kMaxPairs / 16, kHid / 128, kM), 128, 0, stream>>>(
        Ht, W1b, b1, counts, lists, Zb, layer);
    ffn2_kernel<<<dim3(kMaxPairs / 16, kEmb / 128, kM), 128, 0, stream>>>(
        Zb, W2b, b2, gateOf, counts, lists, O, layer);
    combine_kernel<<<(kTok * kEmb) / 256, 256, 0, stream>>>(h, O);
  }
}
